// TransformerDecoderBlock_22857815949430
// MI455X (gfx1250) — matrix operand order repaired and checked by reading, not compiled, not run on hardware
//
#include <hip/hip_runtime.h>
#include <hip/hip_bf16.h>

#define B_  4
#define T_  1024
#define TK_ 1024
#define D_  1024
#define H_  16
#define DH_ 64
#define FF_ 4096

typedef __attribute__((ext_vector_type(16))) __bf16 v16bf;
typedef __attribute__((ext_vector_type(8)))  __bf16 v8bf;
typedef __attribute__((ext_vector_type(4)))  __bf16 v4bf;
typedef __attribute__((ext_vector_type(8)))  float  v8f;

// ------------------------------------------------------------- async copy
// 16B global->LDS async copy, GVS form: uniform SGPR base + 32-bit lane offset.
__device__ __forceinline__ void async_cp16s(void* lds, const void* sbase,
                                            unsigned voff) {
  unsigned l = (unsigned)(uintptr_t)lds;  // low 32b = LDS offset
  asm volatile("global_load_async_to_lds_b128 %0, %1, %2"
               :: "v"(l), "v"(voff), "s"((unsigned long long)(uintptr_t)sbase)
               : "memory");
}
__device__ __forceinline__ void wait_async0() {
  asm volatile("s_wait_asynccnt 0" ::: "memory");
}

// ------------------------------------------------------------- fragments
__device__ __forceinline__ v16bf frag16(const __bf16* p0, const __bf16* p1) {
  v8bf a = *(const v8bf*)p0;
  v8bf b = *(const v8bf*)p1;
  return __builtin_shufflevector(a, b, 0, 1, 2, 3, 4, 5, 6, 7,
                                 8, 9, 10, 11, 12, 13, 14, 15);
}
// A-operand (16x32): lane m=lane%16; elems 0..7 -> K=g*8+i, 8..15 -> K=16+g*8+i
__device__ __forceinline__ v16bf frag_a(const __bf16* rowp, int g) {
  return frag16(rowp + g * 8, rowp + 16 + g * 8);
}
// B-operand (32x16): lane n=lane%16; elems 0..7 -> K=g*8+i, 8..15 -> K=16+g*8+i
__device__ __forceinline__ v16bf frag_b(const __bf16* colp, int g) {
  return frag16(colp + g * 8, colp + 16 + g * 8);
}

// ------------------------------------------------------------- prepasses
__global__ __launch_bounds__(256) void cvt_bf16_k(const float* __restrict__ in,
                                                  __bf16* __restrict__ out) {
  int i = (blockIdx.x * 256 + threadIdx.x) * 4;
  float4 f = *(const float4*)(in + i);
  v4bf o;
  o[0] = (__bf16)f.x; o[1] = (__bf16)f.y; o[2] = (__bf16)f.z; o[3] = (__bf16)f.w;
  *(v4bf*)(out + i) = o;
}

// W[K][N] fp32 -> Wt[N][K] bf16 (64x64 tiles)
__global__ __launch_bounds__(256) void transp_bf16_k(const float* __restrict__ W,
                                                     __bf16* __restrict__ Wt,
                                                     int K, int N) {
  __shared__ __bf16 tile[64][65];
  const int bx = blockIdx.x * 64;  // N
  const int by = blockIdx.y * 64;  // K
  const int tx = threadIdx.x & 63, ty = threadIdx.x >> 6;
#pragma unroll
  for (int i = 0; i < 64; i += 4)
    tile[ty + i][tx] = (__bf16)W[(size_t)(by + ty + i) * N + bx + tx];
  __syncthreads();
#pragma unroll
  for (int i = 0; i < 64; i += 4)
    Wt[(size_t)(bx + ty + i) * K + by + tx] = tile[tx][ty + i];
}

// ------------------------------------------------------------- GEMM
// C[M,N] = epilogue(A[M,K](bf16) @ Wt[N,K]^T(bf16)); f32 accumulate.
// BK=64: 16 WMMAs per barrier interval, async copies in flight across both
// k-substeps.
#define BM 128
#define BN 128
#define BK 64
#define LDT 72   // padded LDS row (bf16): 144B, 16B aligned + bank spread

#define OUT_F32   0
#define OUT_BF16  1
#define OUT_BF16T 2   // Cb[col*M + row] (pre-transposed V)

template <bool RELU, bool BIAS, bool RES, int OUT>
__global__ __launch_bounds__(256) void gemm_k(
    const __bf16* __restrict__ A, const __bf16* __restrict__ Wt,
    const float* __restrict__ bias, const float* __restrict__ res,
    float* __restrict__ C, __bf16* __restrict__ Cb, int M, int N, int K) {
  __shared__ __bf16 sA[2][BM * LDT];
  __shared__ __bf16 sB[2][BN * LDT];
  const int tid = threadIdx.x;
  const int lane = tid & 31, wave = tid >> 5;
  const int wy = wave >> 1, wx = wave & 1;  // 4 waves over M, 2 over N
  const int g = lane >> 4, l16 = lane & 15;
  const int bm = blockIdx.y * BM, bn = blockIdx.x * BN;

  v8f acc[2][4];
#pragma unroll
  for (int mt = 0; mt < 2; ++mt)
#pragma unroll
    for (int nt = 0; nt < 4; ++nt)
#pragma unroll
      for (int r = 0; r < 8; ++r) acc[mt][nt][r] = 0.0f;

  // staging: 2 threads/row, 64B each per side (4x16B asyncs)
  const int row2 = tid >> 1, ks2 = (tid & 1) * 32;
  const unsigned aOff0 = (unsigned)(((size_t)(bm + row2) * K + ks2) * 2);
  const unsigned bOff0 = (unsigned)(((size_t)(bn + row2) * K + ks2) * 2);

  auto stage = [&](int k0, int buf) {
    __bf16* da = &sA[buf][row2 * LDT + ks2];
    __bf16* db = &sB[buf][row2 * LDT + ks2];
    const unsigned ao = aOff0 + (unsigned)(k0 * 2);
    const unsigned bo = bOff0 + (unsigned)(k0 * 2);
#pragma unroll
    for (int i = 0; i < 4; ++i) {
      async_cp16s(da + 8 * i, A, ao + 16 * i);
      async_cp16s(db + 8 * i, Wt, bo + 16 * i);
    }
  };

  stage(0, 0);
  wait_async0();
  __syncthreads();

  const int nT = K / BK;
  for (int kt = 0; kt < nT; ++kt) {
    const int cur = kt & 1;
    if (kt + 1 < nT) stage((kt + 1) * BK, cur ^ 1);  // in flight during WMMAs

#pragma unroll
    for (int kk = 0; kk < 2; ++kk) {  // two K=32 substeps per buffer
      v16bf af[2], bbf[4];
#pragma unroll
      for (int mt = 0; mt < 2; ++mt)
        af[mt] = frag_a(&sA[cur][(wy * 32 + mt * 16 + l16) * LDT + kk * 32], g);
#pragma unroll
      for (int nt = 0; nt < 4; ++nt)
        bbf[nt] = frag_b(&sB[cur][(wx * 64 + nt * 16 + l16) * LDT + kk * 32], g);
#pragma unroll
      for (int mt = 0; mt < 2; ++mt)
#pragma unroll
        for (int nt = 0; nt < 4; ++nt)
          acc[mt][nt] = __builtin_amdgcn_wmma_f32_16x16x32_bf16(
              false, af[mt], false, bbf[nt], (short)0, acc[mt][nt], false,
              false);
    }

    if (kt + 1 < nT) {
      wait_async0();
      __syncthreads();
    }
  }

  // epilogue: C/D layout — VGPR r holds row r+8*g, col = l16
#pragma unroll
  for (int mt = 0; mt < 2; ++mt)
#pragma unroll
    for (int nt = 0; nt < 4; ++nt) {
      int col = bn + wx * 64 + nt * 16 + l16;
      float bv = BIAS ? bias[col] : 0.0f;
#pragma unroll
      for (int r = 0; r < 8; ++r) {
        int row = bm + wy * 32 + mt * 16 + r + 8 * g;
        float v = acc[mt][nt][r];
        if (BIAS) v += bv;
        if (RES) v += res[(size_t)row * N + col];
        if (RELU) v = fmaxf(v, 0.0f);
        if (OUT == OUT_F32) C[(size_t)row * N + col] = v;
        else if (OUT == OUT_BF16) Cb[(size_t)row * N + col] = (__bf16)v;
        else Cb[(size_t)col * M + row] = (__bf16)v;
      }
    }
}

// ------------------------------------------------------------- attention
// Block = (b, h, 64-query tile); 4 waves x 16 query rows. All-bf16 operands.
// Vt layout: [(h*64+dh)][b*TK + key]. MODE 0: causal; MODE 1: per-batch len.
template <int MODE>
__global__ __launch_bounds__(128) void attn_k(
    const __bf16* __restrict__ Qg, const __bf16* __restrict__ Kg,
    const __bf16* __restrict__ Vt, const int* __restrict__ validLens,
    __bf16* __restrict__ Og) {
  constexpr float SCALE = 0.125f;       // 1/sqrt(DH)
  constexpr float MASKV = -1000000.0f;  // matches reference
  __shared__ __bf16 sK[2][64 * 72];     // [key][dh]
  __shared__ __bf16 sV[2][64 * 72];     // [dh][key]
  __shared__ __bf16 sP[4 * 16 * 72];    // per-wave P tile [16][64]

  const int qb = blockIdx.x * 64;
  const int h = blockIdx.y, b = blockIdx.z;
  const int tid = threadIdx.x, lane = tid & 31, wave = tid >> 5;
  const int g = lane >> 4, l16 = lane & 15;
  const size_t hoff = (size_t)h * DH_;
  const int qrow0 = qb + wave * 16;

  // persisted Q fragments (bf16 straight from global)
  v16bf qf[2];
  {
    const __bf16* qp = Qg + ((size_t)b * T_ + qrow0 + l16) * D_ + hoff;
    qf[0] = frag_a(qp, g);
    qf[1] = frag_a(qp + 32, g);
  }

  float mrow[8], lrow[8];
  v8f o[4];
#pragma unroll
  for (int r = 0; r < 8; ++r) { mrow[r] = -1e30f; lrow[r] = 0.0f; }
#pragma unroll
  for (int dt = 0; dt < 4; ++dt)
#pragma unroll
    for (int r = 0; r < 8; ++r) o[dt][r] = 0.0f;

  int vb = 0, nCh;
  if (MODE == 0) nCh = (qb >> 6) + 1;               // causal: skip future
  else { vb = validLens[b]; nCh = (vb + 63) >> 6; } // skip fully-masked keys

  __bf16* sPw = sP + wave * 16 * 72;

  // async chunk staging: rowi = key (K) and dh (V); 64B per thread each side
  const int rowi = tid >> 1, h2 = (tid & 1) * 32;
  const unsigned kOff0 =
      (unsigned)(((size_t)(b * TK_ + rowi) * D_ + hoff + h2) * 2);
  const unsigned vOff0 =
      (unsigned)(((hoff + rowi) * (size_t)(B_ * TK_) + (size_t)b * TK_ + h2) * 2);
  auto stageKV = [&](int kc0, int buf) {
    __bf16* kd = &sK[buf][rowi * 72 + h2];
    __bf16* vd = &sV[buf][rowi * 72 + h2];
    const unsigned ko = kOff0 + (unsigned)(kc0 * D_ * 2);
    const unsigned vo = vOff0 + (unsigned)(kc0 * 2);
#pragma unroll
    for (int i = 0; i < 4; ++i) {
      async_cp16s(kd + 8 * i, Kg, ko + 16 * i);
      async_cp16s(vd + 8 * i, Vt, vo + 16 * i);
    }
  };

  stageKV(0, 0);
  wait_async0();
  __syncthreads();

  for (int ch = 0; ch < nCh; ++ch) {
    const int cur = ch & 1;
    const int kc0 = ch * 64;
    if (ch + 1 < nCh) stageKV((ch + 1) * 64, cur ^ 1);  // overlap with compute

    // S = Q @ K^T (K's natural [key][dh] layout IS the B-operand)
    v8f s[4];
#pragma unroll
    for (int nt = 0; nt < 4; ++nt)
#pragma unroll
      for (int r = 0; r < 8; ++r) s[nt][r] = 0.0f;
#pragma unroll
    for (int kk = 0; kk < 2; ++kk)
#pragma unroll
      for (int nt = 0; nt < 4; ++nt) {
        v16bf bk = frag_b(&sK[cur][(nt * 16 + l16) * 72 + kk * 32], g);
        s[nt] = __builtin_amdgcn_wmma_f32_16x16x32_bf16(
            false, qf[kk], false, bk, (short)0, s[nt], false, false);
      }

    // scale + mask + online softmax (row stats across 16 lanes per half)
#pragma unroll
    for (int r = 0; r < 8; ++r) {
      const int qrow = qrow0 + r + 8 * g;
      float mc = -1e30f;
#pragma unroll
      for (int nt = 0; nt < 4; ++nt) {
        int kc = kc0 + nt * 16 + l16;
        float sv = s[nt][r] * SCALE;
        bool msk = (MODE == 0) ? (kc > qrow) : (kc >= vb);
        if (msk) sv = MASKV;
        s[nt][r] = sv;
        mc = fmaxf(mc, sv);
      }
#pragma unroll
      for (int i = 1; i < 16; i <<= 1) mc = fmaxf(mc, __shfl_xor(mc, i, 32));
      float mn = fmaxf(mrow[r], mc);
      float alpha = __expf(mrow[r] - mn);
      float rs = 0.0f;
#pragma unroll
      for (int nt = 0; nt < 4; ++nt) {
        float p = __expf(s[nt][r] - mn);
        rs += p;
        sPw[(r + 8 * g) * 72 + nt * 16 + l16] = (__bf16)p;  // C-layout -> LDS
      }
#pragma unroll
      for (int i = 1; i < 16; i <<= 1) rs += __shfl_xor(rs, i, 32);
      lrow[r] = lrow[r] * alpha + rs;
      mrow[r] = mn;
#pragma unroll
      for (int dt = 0; dt < 4; ++dt) o[dt][r] *= alpha;
    }

    // O += P @ V  (P from LDS in A-layout; V^T gives B-layout)
#pragma unroll
    for (int kk = 0; kk < 2; ++kk) {
      v16bf ap = frag_a(sPw + l16 * 72 + kk * 32, g);
#pragma unroll
      for (int dt = 0; dt < 4; ++dt) {
        v16bf bv = frag_b(&sV[cur][(dt * 16 + l16) * 72 + kk * 32], g);
        o[dt] = __builtin_amdgcn_wmma_f32_16x16x32_bf16(
            false, ap, false, bv, (short)0, o[dt], false, false);
      }
    }
    wait_async0();
    __syncthreads();
  }

  // normalize, write bf16 in [b, q, h*DH + dh] layout
#pragma unroll
  for (int r = 0; r < 8; ++r) {
    float inv = 1.0f / lrow[r];
    const int qrow = qrow0 + r + 8 * g;
    __bf16* op = Og + ((size_t)b * T_ + qrow) * D_ + hoff;
#pragma unroll
    for (int dt = 0; dt < 4; ++dt) op[dt * 16 + l16] = (__bf16)(o[dt][r] * inv);
  }
}

// ------------------------------------------------------------- LayerNorm
__global__ __launch_bounds__(256) void ln_k(
    const float* __restrict__ x, const float* __restrict__ gma,
    const float* __restrict__ bta, float* __restrict__ out,
    __bf16* __restrict__ outb) {
  __shared__ float r1[8], r2[8];
  const int row = blockIdx.x, tid = threadIdx.x;
  const float* xr = x + (size_t)row * D_;
  float4 v = ((const float4*)xr)[tid];
  float s = v.x + v.y + v.z + v.w;
  float s2 = v.x * v.x + v.y * v.y + v.z * v.z + v.w * v.w;
#pragma unroll
  for (int i = 1; i < 32; i <<= 1) {
    s += __shfl_xor(s, i, 32);
    s2 += __shfl_xor(s2, i, 32);
  }
  if ((tid & 31) == 0) { r1[tid >> 5] = s; r2[tid >> 5] = s2; }
  __syncthreads();
  float ts = 0.0f, ts2 = 0.0f;
#pragma unroll
  for (int i = 0; i < 8; ++i) { ts += r1[i]; ts2 += r2[i]; }
  const float mean = ts * (1.0f / D_);
  const float var = ts2 * (1.0f / D_) - mean * mean;
  const float rstd = rsqrtf(var + 1e-5f);
  float4 gv = ((const float4*)gma)[tid];
  float4 bv = ((const float4*)bta)[tid];
  float4 ov;
  ov.x = (v.x - mean) * rstd * gv.x + bv.x;
  ov.y = (v.y - mean) * rstd * gv.y + bv.y;
  ov.z = (v.z - mean) * rstd * gv.z + bv.z;
  ov.w = (v.w - mean) * rstd * gv.w + bv.w;
  ((float4*)(out + (size_t)row * D_))[tid] = ov;
  if (outb) {
    v4bf ob;
    ob[0] = (__bf16)ov.x; ob[1] = (__bf16)ov.y;
    ob[2] = (__bf16)ov.z; ob[3] = (__bf16)ov.w;
    *(v4bf*)(outb + (size_t)row * D_ + tid * 4) = ob;
  }
}

// ------------------------------------------------------------- launch
extern "C" void kernel_launch(void* const* d_in, const int* in_sizes, int n_in,
                              void* d_out, int out_size, void* d_ws,
                              size_t ws_size, hipStream_t stream) {
  const float* X   = (const float*)d_in[0];
  const float* E   = (const float*)d_in[1];
  const int* encVL = (const int*)d_in[3];
  const float* Wf[10] = {(const float*)d_in[4],  (const float*)d_in[5],
                         (const float*)d_in[6],  (const float*)d_in[7],
                         (const float*)d_in[8],  (const float*)d_in[9],
                         (const float*)d_in[10], (const float*)d_in[11],
                         (const float*)d_in[18], (const float*)d_in[20]};
  const float* g1  = (const float*)d_in[12];
  const float* be1 = (const float*)d_in[13];
  const float* g2  = (const float*)d_in[14];
  const float* be2 = (const float*)d_in[15];
  const float* g3  = (const float*)d_in[16];
  const float* be3 = (const float*)d_in[17];
  const float* bb1 = (const float*)d_in[19];
  const float* bb2 = (const float*)d_in[21];

  const size_t MT = (size_t)B_ * T_ * D_;  // 4M elements
  char* p = (char*)d_ws;
  auto abf = [&](size_t n) { __bf16* r = (__bf16*)p; p += n * 2; return r; };
  auto af32 = [&](size_t n) { float* r = (float*)p; p += n * 4; return r; };

  __bf16* Xbf = abf(MT);
  __bf16* Ebf = abf(MT);
  __bf16* Wt[10];
  for (int i = 0; i < 8; ++i) Wt[i] = abf((size_t)D_ * D_);
  Wt[8] = abf((size_t)D_ * FF_);   // W1t [FF][D]
  Wt[9] = abf((size_t)FF_ * D_);   // W2t [D][FF]
  __bf16* qbf  = abf(MT);
  __bf16* kbf  = abf(MT);
  __bf16* vtbf = abf(MT);          // [h*64+dh][b*TK+key]
  __bf16* attb = abf(MT);
  __bf16* Ybf  = abf(MT);
  __bf16* Zbf  = abf(MT);
  __bf16* ffhb = abf((size_t)B_ * T_ * FF_);
  float* Y  = af32(MT);
  float* Z  = af32(MT);
  float* t1 = af32(MT);

  const int M = B_ * T_;
  dim3 blk(256);
  dim3 gD(D_ / BN, M / BM);   // (8, 32)
  dim3 gF(FF_ / BN, M / BM);  // (32, 32)
  dim3 ga(T_ / 64, H_, B_);   // (16, 16, 4)
  dim3 ablk(128);

  // ---- prepass: bf16 conversions + weight transposes
  cvt_bf16_k<<<MT / 1024, blk, 0, stream>>>(X, Xbf);
  cvt_bf16_k<<<MT / 1024, blk, 0, stream>>>(E, Ebf);
  for (int i = 0; i < 8; ++i)
    transp_bf16_k<<<dim3(D_ / 64, D_ / 64), blk, 0, stream>>>(Wf[i], Wt[i], D_, D_);
  transp_bf16_k<<<dim3(FF_ / 64, D_ / 64), blk, 0, stream>>>(Wf[8], Wt[8], D_, FF_);
  transp_bf16_k<<<dim3(D_ / 64, FF_ / 64), blk, 0, stream>>>(Wf[9], Wt[9], FF_, D_);

  // ---- self-attention
  gemm_k<false, false, false, OUT_BF16><<<gD, blk, 0, stream>>>(
      Xbf, Wt[0], nullptr, nullptr, nullptr, qbf, M, D_, D_);
  gemm_k<false, false, false, OUT_BF16><<<gD, blk, 0, stream>>>(
      Xbf, Wt[1], nullptr, nullptr, nullptr, kbf, M, D_, D_);
  gemm_k<false, false, false, OUT_BF16T><<<gD, blk, 0, stream>>>(
      Xbf, Wt[2], nullptr, nullptr, nullptr, vtbf, M, D_, D_);
  attn_k<0><<<ga, ablk, 0, stream>>>(qbf, kbf, vtbf, nullptr, attb);
  gemm_k<false, false, true, OUT_F32><<<gD, blk, 0, stream>>>(
      attb, Wt[3], nullptr, X, t1, nullptr, M, D_, D_);
  ln_k<<<M, blk, 0, stream>>>(t1, g1, be1, Y, Ybf);

  // ---- cross-attention
  gemm_k<false, false, false, OUT_BF16><<<gD, blk, 0, stream>>>(
      Ybf, Wt[4], nullptr, nullptr, nullptr, qbf, M, D_, D_);
  gemm_k<false, false, false, OUT_BF16><<<gD, blk, 0, stream>>>(
      Ebf, Wt[5], nullptr, nullptr, nullptr, kbf, M, D_, D_);
  gemm_k<false, false, false, OUT_BF16T><<<gD, blk, 0, stream>>>(
      Ebf, Wt[6], nullptr, nullptr, nullptr, vtbf, M, D_, D_);
  attn_k<1><<<ga, ablk, 0, stream>>>(qbf, kbf, vtbf, encVL, attb);
  gemm_k<false, false, true, OUT_F32><<<gD, blk, 0, stream>>>(
      attb, Wt[7], nullptr, Y, t1, nullptr, M, D_, D_);
  ln_k<<<M, blk, 0, stream>>>(t1, g2, be2, Z, Zbf);

  // ---- FFN
  gemm_k<true, true, false, OUT_BF16><<<gF, blk, 0, stream>>>(
      Zbf, Wt[8], bb1, nullptr, nullptr, ffhb, M, FF_, D_);
  gemm_k<false, true, true, OUT_F32><<<gD, blk, 0, stream>>>(
      ffhb, Wt[9], bb2, Z, t1, nullptr, M, D_, FF_);
  ln_k<<<M, blk, 0, stream>>>(t1, g3, be3, (float*)d_out, nullptr);
}
